// SpatialAttentionModule_7361573945816
// MI455X (gfx1250) — matrix operand order repaired and checked by reading, not compiled, not run on hardware
//
#include <hip/hip_runtime.h>
#include <hip/hip_bf16.h>

#define HEADS 4
#define B_    16
#define N_    5000
#define H_    512
#define HD    128
#define R_    8
#define NPAD  5024           // N padded to multiple of 32 (bf16 WMMA K step)
#define NSLAB (NPAD / 32)    // 157 K-slabs for the kv GEMM
#define NRB   40             // ceil(NPAD/128) row-slabs for attn kernel
#define BN    (B_ * N_)
#define EPSF  1e-8f

typedef __attribute__((ext_vector_type(16))) __bf16 v16bf;
typedef __attribute__((ext_vector_type(8)))  __bf16 v8bf;
typedef __attribute__((ext_vector_type(8)))  float  v8f;
typedef __attribute__((ext_vector_type(2)))  float  v2f;
typedef __attribute__((ext_vector_type(4)))  unsigned int v4u;
typedef __attribute__((ext_vector_type(8)))  int v8i;
typedef __attribute__((ext_vector_type(4)))  int v4i;

#if __has_builtin(__builtin_amdgcn_tensor_load_to_lds) && __has_builtin(__builtin_amdgcn_s_wait_tensorcnt)
#define HAS_TDM 1
#else
#define HAS_TDM 0
#endif

__device__ __forceinline__ float elu1(float x) { return x > 0.f ? x + 1.f : expf(x); }

#if HAS_TDM
// Issue a 1-D TDM copy of nelem bf16 elements (global -> LDS).
// D# layout per CDNA5 ISA 8.3/8.4: group0 {count=1, lds_addr, global_addr, type=2},
// group1 {data_size=1(2B), tensor_dim0=nelem, tile_dim0=nelem, dim0_stride=nelem}.
// clang-23 / amdgpu-toolchain form: 6 args (g0, g1, g2, g3, g_ext, cpol).
__device__ __forceinline__ void tdm_load_1d(unsigned int lds_off, const void* gsrc,
                                            unsigned int nelem) {
  unsigned long long ga = (unsigned long long)(uintptr_t)gsrc;
  v4u g0;
  g0[0] = 1u;                                             // count=1, user mode
  g0[1] = lds_off;                                        // lds_addr (bytes)
  g0[2] = (unsigned int)ga;                               // global_addr[31:0]
  g0[3] = (unsigned int)((ga >> 32) & 0x1FFFFFFu) | 0x80000000u;  // addr[56:32] | type=2
  v8i g1;
  g1[0] = (int)(1u << 16);                                // data_size=1 (2 bytes)
  g1[1] = (int)((nelem & 0xFFFFu) << 16);                 // tensor_dim0 lo16 @bits[63:48]
  g1[2] = (int)(((nelem >> 16) & 0xFFFFu) | (1u << 16));  // tensor_dim0 hi16, tensor_dim1=1 lo
  g1[3] = (int)((nelem & 0xFFFFu) << 16);                 // tensor_dim1 hi=0, tile_dim0=nelem
  g1[4] = 0;                                              // tile_dim1=0, tile_dim2=0
  g1[5] = (int)nelem;                                     // tensor_dim0_stride lo32
  g1[6] = 0;                                              // stride hi16, dim1_stride lo16
  g1[7] = 0;
  v4i z; z[0] = 0; z[1] = 0; z[2] = 0; z[3] = 0;
  v8i z8; z8[0] = 0; z8[1] = 0; z8[2] = 0; z8[3] = 0;
  z8[4] = 0; z8[5] = 0; z8[6] = 0; z8[7] = 0;
  __builtin_amdgcn_tensor_load_to_lds(g0, g1, z, z, z8, 0);
}
#endif

// ---- WMMA fragment helpers (bf16 operands use contiguous k = 16*hi + s) ----------------

// A-matrix 16x32 bf16 from row-major [*, lda]: lane m=L&15, hi=L>>4.
__device__ __forceinline__ v16bf load_a16x32(const __bf16* base, int lda, int m0, int k0) {
  const int lane = threadIdx.x & 31;
  const int m = lane & 15, hi = lane >> 4;
  const __bf16* p = base + (size_t)(m0 + m) * lda + k0 + 16 * hi;
  v8bf lo = *(const v8bf*)p;
  v8bf hb = *(const v8bf*)(p + 8);
  v16bf a;
#pragma unroll
  for (int i = 0; i < 8; ++i) { a[i] = lo[i]; a[8 + i] = hb[i]; }
  return a;
}

// B-matrix 32x16 bf16 from packed fragment-major storage (per-lane contiguous).
__device__ __forceinline__ v16bf load_b_packed(const __bf16* base, int tile) {
  const int lane = threadIdx.x & 31;
  const __bf16* p = base + ((size_t)tile * 32 + lane) * 16;
  v8bf lo = *(const v8bf*)p;
  v8bf hb = *(const v8bf*)(p + 8);
  v16bf b;
#pragma unroll
  for (int i = 0; i < 8; ++i) { b[i] = lo[i]; b[8 + i] = hb[i]; }
  return b;
}

__device__ __forceinline__ v8f wmma_bf16(v16bf a, v16bf b, v8f c) {
  return __builtin_amdgcn_wmma_f32_16x16x32_bf16(false, a, false, b, (short)0, c, false, false);
}
__device__ __forceinline__ v8f wmma_f32k4(v2f a, v2f b, v8f c) {
  return __builtin_amdgcn_wmma_f32_16x16x4_f32(false, a, false, b, (short)0, c, false, false);
}

// packed index for B-operand element (k within 32-slab, col within 128)
__device__ __forceinline__ int packB_idx(int tileBase, int kl, int col) {
  int lane = (col & 15) + 16 * (kl >> 4);
  return ((tileBase + (col >> 4)) * 32 + lane) * 16 + (kl & 15);
}

// 8 pipelined WMMAs over one packed 32x128 slab
__device__ __forceinline__ void wmma_slab(const __bf16* buf, int tileBase, v16bf a, v8f c[8]) {
  v16bf bcur = load_b_packed(buf, tileBase + 0);
#pragma unroll
  for (int ct = 0; ct < 7; ++ct) {
    v16bf bnext = load_b_packed(buf, tileBase + ct + 1);
    c[ct] = wmma_bf16(a, bcur, c[ct]);
    bcur = bnext;
  }
  c[7] = wmma_bf16(a, bcur, c[7]);
}

// ---- 1) skinny GEMM: Y[row,j] = bias[j] + X[row,:K] . W[j,:K] ------------
__global__ void k_skinny(const float* __restrict__ X, const float* __restrict__ W,
                         const float* __restrict__ bias, float* __restrict__ Y,
                         int K, int J) {
  long long idx = (long long)blockIdx.x * blockDim.x + threadIdx.x;
  long long total = (long long)BN * J;
  if (idx >= total) return;
  long long row = idx / J;
  int j = (int)(idx % J);
  const float4* x4 = (const float4*)(X + row * (long long)K);
  const float4* w4 = (const float4*)(W + (long long)j * K);
  float acc = bias[j];
  int k4 = K >> 2;
  for (int k = 0; k < k4; ++k) {
    float4 a = x4[k], b = w4[k];
    acc += a.x * b.x + a.y * b.y + a.z * b.z + a.w * b.w;
  }
  Y[idx] = acc;
}

// ---- 2) high QKV projection + ELU+1 + head-split + bf16 staging ----------
__global__ void k_qkv_high(const float* __restrict__ t0, const float* __restrict__ Wh,
                           const float* __restrict__ bqh, __bf16* __restrict__ qf,
                           __bf16* __restrict__ kfT, __bf16* __restrict__ vpk,
                           float* __restrict__ zinv) {
  const int b = blockIdx.x / NPAD;
  const int n = blockIdx.x % NPAD;
  const int t = threadIdx.x;
  const int s = n >> 5, kl = n & 31;
  if (n >= N_) {  // zero-fill K-dim padding so WMMA sums are exact
#pragma unroll
    for (int i = 0; i < 2; ++i) {
      int j = t + 256 * i;             // 0..511
      int h = j >> 7, d = j & 127;
      size_t bh = (size_t)(b * 4 + h);
      qf[(bh * NPAD + n) * HD + d] = (__bf16)0.f;
      kfT[(bh * HD + d) * NPAD + n] = (__bf16)0.f;
      vpk[bh * NPAD * HD + packB_idx(s * 8, kl, d)] = (__bf16)0.f;
    }
    return;
  }
  __shared__ float st[24];
  __shared__ float zs[4];
  if (t < 24) st[t] = t0[((size_t)b * N_ + n) * 24 + t];
  if (t < 4) zs[t] = 0.f;
  __syncthreads();
#pragma unroll
  for (int jj = 0; jj < 6; ++jj) {
    int j = t * 6 + jj;                // 0..1535
    float acc = bqh[j];
    const float* w = Wh + (size_t)j * 24;
#pragma unroll
    for (int c = 0; c < 24; ++c) acc += st[c] * w[c];
    int part = j >> 9;                 // 0=q 1=k 2=v
    int jm = j & 511;
    int h = jm >> 7, d = jm & 127;
    size_t bh = (size_t)(b * 4 + h);
    if (part == 0) {
      qf[(bh * NPAD + n) * HD + d] = (__bf16)elu1(acc);
    } else if (part == 1) {
      float kf = elu1(acc);
      kfT[(bh * HD + d) * NPAD + n] = (__bf16)kf;
      atomicAdd(&zs[h], kf);
    } else {
      vpk[bh * NPAD * HD + packB_idx(s * 8, kl, d)] = (__bf16)acc;
    }
  }
  __syncthreads();
  if (t < 4) zinv[(size_t)(b * 4 + t) * N_ + n] = 1.f / (zs[t] + EPSF);
}

// ---- 3) kv[d,e] = sum_n kfT[d,n] * v[n,e]  (WMMA over NPAD) --------------
// TDM double-buffers the packed v slabs into LDS while WMMA consumes.
__global__ void k_kv(const __bf16* __restrict__ kfT, const __bf16* __restrict__ vpk,
                     __bf16* __restrict__ kvp) {
  const int bh = blockIdx.x;           // 0..63
  const int t = threadIdx.x;
  const int wave = t >> 5, lane = t & 31;
  const __bf16* A  = kfT + (size_t)bh * HD * NPAD;
  const __bf16* Bg = vpk + (size_t)bh * NPAD * HD;
  __shared__ __align__(16) __bf16 sB[2][32 * HD];   // 2 x 8KB ping-pong
  v8f c[8] = {};
  const int m0 = wave * 16;
  for (int s = 0; s < NSLAB; ++s) {
#if HAS_TDM
    if (wave == 0) {
      if (s == 0)
        tdm_load_1d((unsigned int)(uintptr_t)&sB[0][0], Bg, 32 * HD);
      if (s + 1 < NSLAB) {
        tdm_load_1d((unsigned int)(uintptr_t)&sB[(s + 1) & 1][0],
                    Bg + (size_t)(s + 1) * 32 * HD, 32 * HD);
        __builtin_amdgcn_s_wait_tensorcnt(1);   // slab s complete, s+1 in flight
      } else {
        __builtin_amdgcn_s_wait_tensorcnt(0);
      }
    }
    __syncthreads();
#else
    {
      const uint32_t* src = (const uint32_t*)(Bg + (size_t)s * 32 * HD);
      uint32_t* dst = (uint32_t*)&sB[s & 1][0];
#pragma unroll
      for (int i = 0; i < 8; ++i) dst[t + 256 * i] = src[t + 256 * i];
    }
    __syncthreads();
    if (s + 1 < NSLAB) __builtin_prefetch(Bg + (size_t)(s + 1) * 32 * HD, 0, 1);
#endif
    v16bf a = load_a16x32(A, NPAD, m0, s * 32);
    wmma_slab(&sB[s & 1][0], 0, a, c);
    __syncthreads();                   // buffer s&1 free for reuse (TDM two iters later)
  }
  const int nn = lane & 15, hi = lane >> 4;
  __bf16* D = kvp + (size_t)bh * (HD * HD);
#pragma unroll
  for (int ct = 0; ct < 8; ++ct)
#pragma unroll
    for (int i = 0; i < 8; ++i) {
      int row = m0 + i + 8 * hi;       // d (K of attn GEMM)
      int col = ct * 16 + nn;          // e (col of attn GEMM)
      D[packB_idx((row >> 5) * 8, row & 31, col)] = (__bf16)c[ct][i];
    }
}

// ---- 4) vsum[h,r] = sum_n elu1(v_param[h,r,n]) ---------------------------
__global__ void k_vsum(const float* __restrict__ v_param, float* __restrict__ vsum) {
  const int hr = blockIdx.x;
  const int t = threadIdx.x;
  __shared__ float sred[256];
  float acc = 0.f;
  for (int n = t; n < N_; n += 256) acc += elu1(v_param[(size_t)hr * N_ + n]);
  sred[t] = acc; __syncthreads();
  for (int s = 128; s > 0; s >>= 1) { if (t < s) sred[t] += sred[t + s]; __syncthreads(); }
  if (t == 0) vsum[hr] = sred[0];
}

// ---- 4b) vT[h,n,r] = v_param[h,r,n]  (for f32 WMMA B-fragments) ----------
__global__ void k_vt(const float* __restrict__ v_param, float* __restrict__ vT) {
  int idx = blockIdx.x * blockDim.x + threadIdx.x;
  if (idx >= HEADS * R_ * N_) return;
  int h = idx / (R_ * N_);
  int r = (idx / N_) % R_;
  int n = idx % N_;
  vT[((size_t)h * N_ + n) * R_ + r] = v_param[idx];
}

// ---- 5) tmp[b,h,r,d] = sum_n elu1(v_param[h,r,n]) * v[b,h,n,d] -----------
__global__ void k_tmp(const float* __restrict__ v_param, const __bf16* __restrict__ vpk,
                      float* __restrict__ tmpb) {
  const int bh = blockIdx.x;
  const int h = bh & 3;
  const int t = threadIdx.x;
  __shared__ float sv[R_ * 128];
  const int d = t & 127, rg = t >> 7;  // rg 0..1 -> r = rg*4 + rr
  const int ctd = (d >> 4) * 32 * 16, dl = d & 15;
  const __bf16* Vb = vpk + (size_t)bh * NPAD * HD;
  float acc[4] = {0.f, 0.f, 0.f, 0.f};
  for (int n0 = 0; n0 < N_; n0 += 128) {
#pragma unroll
    for (int i = 0; i < 4; ++i) {
      int idx = t + 256 * i;           // == r*128 + nn
      int r = idx >> 7, nn = idx & 127;
      int n = n0 + nn;
      sv[idx] = (n < N_) ? elu1(v_param[((size_t)h * R_ + r) * N_ + n]) : 0.f;
    }
    __syncthreads();
    int lim = (N_ - n0 < 128) ? (N_ - n0) : 128;
    for (int nn = 0; nn < lim; ++nn) {
      int n = n0 + nn, kl = n & 31;
      size_t addr = (size_t)(n >> 5) * (32 * HD) + ctd +
                    ((dl + 16 * (kl >> 4)) * 16) + (kl & 15);
      float vv = (float)Vb[addr];
#pragma unroll
      for (int rr = 0; rr < 4; ++rr) acc[rr] += sv[(rg * 4 + rr) * 128 + nn] * vv;
    }
    __syncthreads();
  }
#pragma unroll
  for (int rr = 0; rr < 4; ++rr)
    tmpb[((size_t)bh * R_ + rg * 4 + rr) * HD + d] = acc[rr];
}

// ---- 6) attn = (qf @ kv)*z + graph_bias -> d_out (merged heads) ----------
__global__ void k_attn(const __bf16* __restrict__ qf, const __bf16* __restrict__ kvp,
                       const float* __restrict__ zinv, const float* __restrict__ tmpb,
                       const float* __restrict__ u, const float* __restrict__ vsum,
                       float* __restrict__ out) {
  const int bh = blockIdx.x / NRB;
  const int rb = blockIdx.x % NRB;
  const int b = bh >> 2, h = bh & 3;
  const int t = threadIdx.x;
  __shared__ __align__(16) __bf16 sKV[HD * HD];   // 32 KB packed fragment-major
  __shared__ float sTmp[R_ * HD];
  __shared__ float sUp[128 * R_];
  __shared__ float sRd[128];
  __shared__ float sZ[128];
#if HAS_TDM
  if (t < 32) {                        // wave 0 issues the DMA and waits
    tdm_load_1d((unsigned int)(uintptr_t)&sKV[0], kvp + (size_t)bh * HD * HD, HD * HD);
    __builtin_amdgcn_s_wait_tensorcnt(0);
  }
#else
  {
    const uint32_t* src = (const uint32_t*)(kvp + (size_t)bh * HD * HD);
    uint32_t* dst = (uint32_t*)sKV;
#pragma unroll
    for (int i = 0; i < 32; ++i) dst[t + 256 * i] = src[t + 256 * i];
  }
#endif
#pragma unroll
  for (int i = 0; i < 4; ++i) sTmp[t + 256 * i] = tmpb[(size_t)bh * (R_ * HD) + t + 256 * i];
#pragma unroll
  for (int i = 0; i < 4; ++i) {
    int idx = t + 256 * i;             // 0..1023 == nl*8 + r
    int nl = idx >> 3, r = idx & 7;
    int n = rb * 128 + nl;
    sUp[idx] = (n < N_) ? elu1(u[((size_t)h * N_ + n) * R_ + r]) : 0.f;
  }
  if (t < 128) {
    int n = rb * 128 + t;
    sZ[t] = (n < N_) ? zinv[(size_t)bh * N_ + n] : 0.f;
  }
  __syncthreads();
  if (t < 128) {
    float dn = 0.f;
#pragma unroll
    for (int r = 0; r < R_; ++r) dn += sUp[t * R_ + r] * vsum[h * R_ + r];
    sRd[t] = 1.f / (dn + EPSF);
  }
  __syncthreads();
  const int wave = t >> 5, lane = t & 31;
  const int m0g = rb * 128 + wave * 16;
  if (m0g >= NPAD) return;
  const __bf16* Aq = qf + (size_t)bh * NPAD * HD;
  v8f c[8] = {};
#pragma unroll
  for (int kk = 0; kk < 4; ++kk) {
    v16bf a = load_a16x32(Aq, HD, m0g, kk * 32);
    wmma_slab(sKV, kk * 8, a, c);
  }
  const int nn = lane & 15, hi = lane >> 4;
#pragma unroll
  for (int ct = 0; ct < 8; ++ct)
#pragma unroll
    for (int i = 0; i < 8; ++i) {
      int rl = wave * 16 + i + 8 * hi;  // local row in slab
      int n = rb * 128 + rl;
      int col = ct * 16 + nn;
      float val = c[ct][i] * sZ[rl];
      float g = 0.f;
#pragma unroll
      for (int r = 0; r < R_; ++r) g += sUp[rl * R_ + r] * sTmp[r * HD + col];
      val += g * sRd[rl];
      if (n < N_) out[((size_t)b * N_ + n) * H_ + h * HD + col] = val;
    }
}

// ---- 7) d_out += out_proj_high(tlow) -------------------------------------
__global__ void k_final(const float* __restrict__ tlow, const float* __restrict__ Woh,
                        const float* __restrict__ boh, float* __restrict__ out) {
  size_t idx = (size_t)blockIdx.x * blockDim.x + threadIdx.x;
  if (idx >= (size_t)BN * H_) return;
  size_t row = idx >> 9;
  int j = (int)(idx & 511);
  const float* tl = tlow + row * 8;
  const float* w = Woh + (size_t)j * 8;
  float s = boh[j];
#pragma unroll
  for (int c = 0; c < 8; ++c) s += tl[c] * w[c];
  out[idx] += s;
}

// ---- 8) reg loss = exp(lw) * mean |u @ v_param|  (f32 WMMA 16x16x4) ------
__global__ void k_regzero(float* out) {
  if (threadIdx.x == 0 && blockIdx.x == 0) out[(size_t)BN * H_] = 0.f;
}

__global__ void k_reg(const float* __restrict__ u, const float* __restrict__ vT,
                      const float* __restrict__ logw, float* __restrict__ out) {
  const int nt = (N_ + 15) / 16;       // 313
  const int h = blockIdx.x / nt;
  const int it = blockIdx.x % nt;
  const int t = threadIdx.x;
  const int wave = t >> 5, lane = t & 31;
  const int m = lane & 15, hi = lane >> 4;
  const int row = it * 16 + m;
  v2f alo, ahi;
  if (row < N_) {
    const float* up = u + ((size_t)h * N_ + row) * R_;
    alo = *(const v2f*)(up + 2 * hi);          // A elem e -> K = e + 2*hi
    ahi = *(const v2f*)(up + 4 + 2 * hi);
  } else {
    alo[0] = alo[1] = ahi[0] = ahi[1] = 0.f;
  }
  float partial = 0.f;
  for (int jt = wave; jt < nt; jt += 8) {
    const int col = jt * 16 + m;
    v2f blo, bhi;
    if (col < N_) {
      const float* vp = vT + ((size_t)h * N_ + col) * R_;
      blo = *(const v2f*)(vp + 2 * hi);        // B elem e -> K = e + 2*hi
      bhi = *(const v2f*)(vp + 4 + 2 * hi);
    } else {
      blo[0] = blo[1] = bhi[0] = bhi[1] = 0.f;
    }
    v8f c = {};
    c = wmma_f32k4(alo, blo, c);       // K = 0..3
    c = wmma_f32k4(ahi, bhi, c);       // K = 4..7
#pragma unroll
    for (int i = 0; i < 8; ++i) partial += fabsf(c[i]);
  }
  __shared__ float sred[256];
  sred[t] = partial; __syncthreads();
  for (int s = 128; s > 0; s >>= 1) { if (t < s) sred[t] += sred[t + s]; __syncthreads(); }
  if (t == 0) {
    float scale = expf(logw[0]) / ((float)HEADS * (float)N_ * (float)N_);
    atomicAdd(out + (size_t)BN * H_, scale * sred[0]);
  }
}

// --------------------------------------------------------------------------
extern "C" void kernel_launch(void* const* d_in, const int* in_sizes, int n_in,
                              void* d_out, int out_size, void* d_ws, size_t ws_size,
                              hipStream_t stream) {
  const float* x       = (const float*)d_in[0];
  const float* Wql     = (const float*)d_in[1];
  const float* bql     = (const float*)d_in[2];
  const float* Wqh     = (const float*)d_in[3];
  const float* bqh     = (const float*)d_in[4];
  const float* Wol     = (const float*)d_in[5];
  const float* bol     = (const float*)d_in[6];
  const float* Woh     = (const float*)d_in[7];
  const float* boh     = (const float*)d_in[8];
  const float* u       = (const float*)d_in[9];
  const float* v_param = (const float*)d_in[10];
  const float* logw    = (const float*)d_in[11];
  float* out = (float*)d_out;
  (void)in_sizes; (void)n_in; (void)out_size; (void)ws_size;

  char* ws = (char*)d_ws;
  size_t off = 0;
  auto take = [&](size_t bytes) -> char* {
    char* p = ws + off;
    off = (off + bytes + 255) & ~(size_t)255;
    return p;
  };
  float*  t0   = (float*)take((size_t)BN * 24 * sizeof(float));
  __bf16* qf   = (__bf16*)take((size_t)B_ * HEADS * NPAD * HD * 2);
  __bf16* kfT  = (__bf16*)take((size_t)B_ * HEADS * HD * NPAD * 2);
  __bf16* vpk  = (__bf16*)take((size_t)B_ * HEADS * NPAD * HD * 2);
  float*  zinv = (float*)take((size_t)B_ * HEADS * N_ * sizeof(float));
  __bf16* kvp  = (__bf16*)take((size_t)B_ * HEADS * HD * HD * 2);
  float*  vsum = (float*)take(32 * sizeof(float));
  float*  vT   = (float*)take((size_t)HEADS * N_ * R_ * sizeof(float));
  float*  tmpb = (float*)take((size_t)B_ * HEADS * R_ * HD * sizeof(float));
  float*  tlow = (float*)take((size_t)BN * 8 * sizeof(float));

  k_skinny<<<(BN * 24 + 255) / 256, 256, 0, stream>>>(x, Wql, bql, t0, 512, 24);
  k_qkv_high<<<B_ * NPAD, 256, 0, stream>>>(t0, Wqh, bqh, qf, kfT, vpk, zinv);
  k_kv<<<B_ * HEADS, 256, 0, stream>>>(kfT, vpk, kvp);
  k_vsum<<<HEADS * R_, 256, 0, stream>>>(v_param, vsum);
  k_vt<<<(HEADS * R_ * N_ + 255) / 256, 256, 0, stream>>>(v_param, vT);
  k_tmp<<<B_ * HEADS, 256, 0, stream>>>(v_param, vpk, tmpb);
  k_attn<<<B_ * HEADS * NRB, 256, 0, stream>>>(qf, kvp, zinv, tmpb, u, vsum, out);
  k_skinny<<<(BN * 8 + 255) / 256, 256, 0, stream>>>(out, Wol, bol, tlow, 512, 8);
  k_final<<<(int)(((size_t)BN * H_ + 255) / 256), 256, 0, stream>>>(tlow, Woh, boh, out);
  k_regzero<<<1, 1, 0, stream>>>(out);
  k_reg<<<HEADS * ((N_ + 15) / 16), 256, 0, stream>>>(u, vT, logw, out);
}
